// Trasnsformer_model_40518721470946
// MI455X (gfx1250) — matrix operand order repaired and checked by reading, not compiled, not run on hardware
//
#include <hip/hip_runtime.h>
#include <hip/hip_bf16.h>

typedef __attribute__((ext_vector_type(16))) _Float16 v16h;
typedef __attribute__((ext_vector_type(8)))  float    v8f;

#define L_ROWS  65536
#define D_IN    256
#define DH      128
#define LEAKC   0.01f
#define BN_EPS  1e-5f
#define CD_DC   (-1.702f)
#define NCHUNK  256
#define CHUNK   256
#define MROWS   64          // rows per GEMM block (4 row-tiles of 16)

// ---------------------------------------------------------------------------
// Column mean/var of I = [Q | C] (two-pass, deterministic, no atomics)
// ---------------------------------------------------------------------------
__global__ __launch_bounds__(256) void k_colstats_partial(
    const float* __restrict__ Q, const float* __restrict__ C,
    float* __restrict__ pS, float* __restrict__ pS2) {
  int b = blockIdx.x, c = threadIdx.x;
  const float* src = (c < DH) ? (Q + c) : (C + (c - DH));
  long base = (long)b * CHUNK;
  float s = 0.f, s2 = 0.f;
  for (int r = 0; r < CHUNK; ++r) {
    float x = src[(base + r) * DH];
    s += x; s2 += x * x;
  }
  pS [b * D_IN + c] = s;
  pS2[b * D_IN + c] = s2;
}

__global__ __launch_bounds__(256) void k_colstats_final(
    const float* __restrict__ pS, const float* __restrict__ pS2,
    float* __restrict__ meanv, float* __restrict__ rstdv) {
  int c = threadIdx.x;
  float s = 0.f, s2 = 0.f;
  for (int b = 0; b < NCHUNK; ++b) { s += pS[b * D_IN + c]; s2 += pS2[b * D_IN + c]; }
  float m = s / (float)L_ROWS;
  float v = s2 / (float)L_ROWS - m * m;
  meanv[c] = m;
  rstdv[c] = rsqrtf(v + BN_EPS);
}

// XN = (I - mean) * rstd, stored f16 row-major (L x 256)
__global__ __launch_bounds__(256) void k_xnorm(
    const float* __restrict__ Q, const float* __restrict__ C,
    const float* __restrict__ meanv, const float* __restrict__ rstdv,
    _Float16* __restrict__ XN) {
  int r = blockIdx.x, c = threadIdx.x;
  float x = (c < DH) ? Q[(size_t)r * DH + c] : C[(size_t)r * DH + (c - DH)];
  XN[(size_t)r * D_IN + c] = (_Float16)((x - meanv[c]) * rstdv[c]);
}

// ---------------------------------------------------------------------------
// Weight prep: fold BN affine into GEMM1; transpose to col-major f16 so each
// WMMA B-fragment lane reads contiguous K pairs.
// ---------------------------------------------------------------------------
__global__ __launch_bounds__(256) void k_prep_w1(
    const float* __restrict__ aw1, const float* __restrict__ g,
    const float* __restrict__ bcol, const float* __restrict__ ab1,
    _Float16* __restrict__ W1t, float* __restrict__ bias1) {
  __shared__ float red[256];
  int h = blockIdx.x, k = threadIdx.x;
  float v = aw1[k * D_IN + h];
  W1t[(size_t)h * D_IN + k] = (_Float16)(v * g[k]);
  red[k] = v * bcol[k];
  for (int s = 128; s > 0; s >>= 1) {
    __syncthreads();
    if (k < s) red[k] += red[k + s];
  }
  __syncthreads();
  if (k == 0) bias1[h] = ab1[h] + red[0];
}

__global__ __launch_bounds__(256) void k_transpose_f16(
    const float* __restrict__ W, _Float16* __restrict__ Wt) {
  int n = blockIdx.x, k = threadIdx.x;
  Wt[(size_t)n * D_IN + k] = (_Float16)W[(size_t)k * D_IN + n];
}

// ---------------------------------------------------------------------------
// Async global -> LDS staging of a 64x256 f16 tile (32 KB) using the gfx1250
// ASYNCcnt path. Per-lane LDS byte address = low 32 bits of the generic
// shared-aperture pointer (ISA 10.2).
// ---------------------------------------------------------------------------
__device__ __forceinline__ void stage_async_32k(
    const _Float16* __restrict__ gsrc, _Float16* lds_buf, int tid) {
  uint32_t lds_base = (uint32_t)(uintptr_t)lds_buf;
  uint64_t gbase = (uint64_t)(uintptr_t)gsrc;
#pragma unroll
  for (int i = 0; i < 8; ++i) {
    uint32_t ldsa = lds_base + (uint32_t)(tid * 16 + i * 4096);
    uint64_t ga   = gbase + (uint64_t)(tid * 16 + i * 4096);
    asm volatile("global_load_async_to_lds_b128 %0, %1, off"
                 :: "v"(ldsa), "v"(ga) : "memory");
  }
  asm volatile("s_wait_asynccnt 0" ::: "memory");
}

// ---------------------------------------------------------------------------
// 64x256 @ 256x256 f16 GEMM micro-block: each wave owns 2 col-tiles; the B
// fragment is held in registers and reused across 4 row-tiles (4x B reuse).
// Layouts per ISA 7.12.2 (A 16x32 / B 32x16 / C,D 16x16 f32).
// ---------------------------------------------------------------------------
__device__ __forceinline__ void wmma_block_64(
    const _Float16* __restrict__ buf,   // LDS [64][256] f16 row-major
    const _Float16* __restrict__ Wt,    // global, Wt[n*256 + k]
    int lane, int wave, v8f acc[2][4]) {
  int row = lane & 15;
  int hk  = lane >> 4;
#pragma unroll
  for (int t = 0; t < 2; ++t) {
    int ct = wave + t * 8;
    const _Float16* wp = Wt + (size_t)(ct * 16 + row) * D_IN;
#pragma unroll
    for (int m = 0; m < 4; ++m) acc[t][m] = {};
    for (int kt = 0; kt < 8; ++kt) {
      int kb = kt * 32;
      v16h b;
#pragma unroll
      for (int j = 0; j < 8; ++j) {
        int kd = kb + (hk ? 16 : 0) + 2 * j;
        b[2 * j]     = wp[kd];
        b[2 * j + 1] = wp[kd + 1];
      }
#pragma unroll
      for (int m = 0; m < 4; ++m) {
        v16h a;
#pragma unroll
        for (int j = 0; j < 8; ++j) {
          int ka = kb + (hk ? 16 : 0) + 2 * j;
          a[2 * j]     = buf[(m * 16 + row) * D_IN + ka];
          a[2 * j + 1] = buf[(m * 16 + row) * D_IN + ka + 1];
        }
        acc[t][m] = __builtin_amdgcn_wmma_f32_16x16x32_f16(
            false, a, false, b, (short)0, acc[t][m], false, false);
      }
    }
  }
}

// bias + leaky, write back to the (already-consumed) LDS buffer as f16
__device__ __forceinline__ void epilogue_to_lds(
    v8f acc[2][4], const float* __restrict__ bias, int lane, int wave,
    _Float16* __restrict__ buf) {
  int ln15 = lane & 15, hk = lane >> 4;
#pragma unroll
  for (int t = 0; t < 2; ++t) {
    int col = (wave + t * 8) * 16 + ln15;
    float bi = bias[col];
#pragma unroll
    for (int m = 0; m < 4; ++m)
#pragma unroll
      for (int r = 0; r < 8; ++r) {
        float v = acc[t][m][r] + bi;
        v = (v >= 0.f) ? v : LEAKC * v;
        buf[(m * 16 + r + (hk ? 8 : 0)) * D_IN + col] = (_Float16)v;
      }
  }
}

// 64 rows x 256 cols (f16 in LDS) dot a 256-vector -> 64 scalars
__device__ __forceinline__ float dot_rows_64(
    const _Float16* __restrict__ buf, const float* __restrict__ w,
    float* __restrict__ Red, int tid) {
  int row = tid >> 2, q = tid & 3;
  float s = 0.f;
#pragma unroll 8
  for (int i = 0; i < 64; ++i)
    s += (float)buf[row * D_IN + q * 64 + i] * w[q * 64 + i];
  Red[tid] = s;
  __syncthreads();
  float r = 0.f;
  if (tid < 64) {
#pragma unroll
    for (int i = 0; i < 4; ++i) r += Red[tid * 4 + i];
  }
  return r;
}

// ---------------------------------------------------------------------------
// att = leaky(XN @ W1' + b1') @ aw2 + ab2   (fused; 64 rows per block)
// ---------------------------------------------------------------------------
__global__ __launch_bounds__(256) void k_att(
    const _Float16* __restrict__ XN, const _Float16* __restrict__ W1t,
    const float* __restrict__ bias1, const float* __restrict__ aw2,
    const float* __restrict__ ab2, float* __restrict__ att) {
  __shared__ __align__(16) _Float16 buf[MROWS * D_IN];
  __shared__ float Red[256];
  int tid = threadIdx.x;
  stage_async_32k(XN + (size_t)blockIdx.x * (MROWS * D_IN), buf, tid);
  __syncthreads();
  int wave = tid >> 5, lane = tid & 31;
  v8f acc[2][4];
  wmma_block_64(buf, W1t, lane, wave, acc);
  __syncthreads();                 // everyone done reading A
  epilogue_to_lds(acc, bias1, lane, wave, buf);
  __syncthreads();
  float s = dot_rows_64(buf, aw2, Red, tid);
  if (tid < 64) att[blockIdx.x * MROWS + tid] = s + ab2[0];
}

// ---------------------------------------------------------------------------
// Chunked inclusive scan of enc = cumsum(att * Q): partials -> scan -> fill TIN
// ---------------------------------------------------------------------------
__global__ __launch_bounds__(128) void k_enc_partial(
    const float* __restrict__ att, const float* __restrict__ Q,
    float* __restrict__ csPart) {
  int b = blockIdx.x, j = threadIdx.x;
  float s = 0.f;
  for (int r = 0; r < CHUNK; ++r) {
    int l = b * CHUNK + r;
    s += att[l] * Q[(size_t)l * DH + j];
  }
  csPart[b * DH + j] = s;
}

__global__ __launch_bounds__(128) void k_enc_scan(float* __restrict__ csPart) {
  int j = threadIdx.x;
  float run = 0.f;
  for (int b = 0; b < NCHUNK; ++b) {
    float t = csPart[b * DH + j];
    csPart[b * DH + j] = run;   // exclusive chunk offset
    run += t;
  }
}

__global__ __launch_bounds__(128) void k_enc_tin(
    const float* __restrict__ att, const float* __restrict__ Q,
    const float* __restrict__ C, const float* __restrict__ csPart,
    _Float16* __restrict__ TIN) {
  int b = blockIdx.x, j = threadIdx.x;
  float run = csPart[b * DH + j];
  for (int r = 0; r < CHUNK; ++r) {
    int l = b * CHUNK + r;
    run += att[l] * Q[(size_t)l * DH + j];          // inclusive cumsum
    TIN[(size_t)l * D_IN + j]      = (_Float16)run;
    TIN[(size_t)l * D_IN + DH + j] = (_Float16)C[(size_t)l * DH + j];
  }
}

// ---------------------------------------------------------------------------
// state = (leaky(leaky(TIN@sw1+b1)@sw2+b2)) @ sw3 + b3, fused; 64 rows/block.
// One 32 KB LDS buffer is recycled: tin -> h1 -> h2 (barrier-separated).
// ---------------------------------------------------------------------------
__global__ __launch_bounds__(256) void k_state(
    const _Float16* __restrict__ TIN,
    const _Float16* __restrict__ S1t, const float* __restrict__ sb1,
    const _Float16* __restrict__ S2t, const float* __restrict__ sb2,
    const float* __restrict__ sw3, const float* __restrict__ sb3,
    float* __restrict__ state) {
  __shared__ __align__(16) _Float16 buf[MROWS * D_IN];
  __shared__ float Red[256];
  int tid = threadIdx.x;
  stage_async_32k(TIN + (size_t)blockIdx.x * (MROWS * D_IN), buf, tid);
  __syncthreads();
  int wave = tid >> 5, lane = tid & 31;
  v8f acc[2][4];
  // layer 1: buf(tin) @ S1t -> buf(h1)
  wmma_block_64(buf, S1t, lane, wave, acc);
  __syncthreads();
  epilogue_to_lds(acc, sb1, lane, wave, buf);
  __syncthreads();
  // layer 2: buf(h1) @ S2t -> buf(h2)
  wmma_block_64(buf, S2t, lane, wave, acc);
  __syncthreads();
  epilogue_to_lds(acc, sb2, lane, wave, buf);
  __syncthreads();
  float s = dot_rows_64(buf, sw3, Red, tid);
  if (tid < 64) state[blockIdx.x * MROWS + tid] = s + sb3[0];
}

// ---------------------------------------------------------------------------
// IRT probability epilogue (u_state is the shifted/offset state sequence)
// ---------------------------------------------------------------------------
__global__ __launch_bounds__(256) void k_prob(
    const float* __restrict__ st1, const float* __restrict__ st2,
    const float* __restrict__ user_init, const float* __restrict__ Qinfo,
    const float* __restrict__ cd_guess, float* __restrict__ p) {
  int l = blockIdx.x * 256 + threadIdx.x;
  float u1 = user_init[0], u2 = user_init[1];
  if (l > 0) { u1 += st1[l - 1]; u2 += st2[l - 1]; }
  float z = (u1 - Qinfo[l * 4 + 0]) * Qinfo[l * 4 + 2]
          + (u2 - Qinfo[l * 4 + 1]) * Qinfo[l * 4 + 3];
  float g = cd_guess[0];
  p[l] = g + (1.f - g) / (1.f + expf(CD_DC * z));
}

// ---------------------------------------------------------------------------
extern "C" void kernel_launch(void* const* d_in, const int* in_sizes, int n_in,
                              void* d_out, int out_size, void* d_ws, size_t ws_size,
                              hipStream_t stream) {
  (void)in_sizes; (void)n_in; (void)out_size; (void)ws_size;
  const float* Q     = (const float*)d_in[0];
  const float* C     = (const float*)d_in[1];
  const float* Qinfo = (const float*)d_in[2];
  const float* br[2][12];
  for (int b = 0; b < 2; ++b)
    for (int j = 0; j < 12; ++j) br[b][j] = (const float*)d_in[3 + b * 12 + j];
  const float* user_init = (const float*)d_in[27];
  const float* cd_guess  = (const float*)d_in[28];
  float* out = (float*)d_out;

  char* ws = (char*)d_ws;
  size_t off = 0;
  auto walloc = [&](size_t bytes) -> void* {
    void* p = ws + off;
    off += (bytes + 255) & ~(size_t)255;
    return p;
  };
  _Float16* XN  = (_Float16*)walloc((size_t)L_ROWS * D_IN * 2);
  _Float16* TIN = (_Float16*)walloc((size_t)L_ROWS * D_IN * 2);
  _Float16* W1t = (_Float16*)walloc((size_t)D_IN * D_IN * 2);
  _Float16* S1t = (_Float16*)walloc((size_t)D_IN * D_IN * 2);
  _Float16* S2t = (_Float16*)walloc((size_t)D_IN * D_IN * 2);
  float* bias1  = (float*)walloc(D_IN * 4);
  float* pS     = (float*)walloc((size_t)NCHUNK * D_IN * 4);
  float* pS2    = (float*)walloc((size_t)NCHUNK * D_IN * 4);
  float* meanv  = (float*)walloc(D_IN * 4);
  float* rstdv  = (float*)walloc(D_IN * 4);
  float* attb   = (float*)walloc((size_t)L_ROWS * 4);
  float* csPart = (float*)walloc((size_t)NCHUNK * DH * 4);
  float* st[2]  = { (float*)walloc((size_t)L_ROWS * 4),
                    (float*)walloc((size_t)L_ROWS * 4) };

  // shared BN stats (mean/var of I is branch-independent; affine folded into W1)
  k_colstats_partial<<<NCHUNK, 256, 0, stream>>>(Q, C, pS, pS2);
  k_colstats_final<<<1, 256, 0, stream>>>(pS, pS2, meanv, rstdv);
  k_xnorm<<<L_ROWS, 256, 0, stream>>>(Q, C, meanv, rstdv, XN);

  for (int b = 0; b < 2; ++b) {
    const float *bg = br[b][0], *bb = br[b][1], *aw1 = br[b][2], *ab1 = br[b][3],
                *aw2 = br[b][4], *ab2 = br[b][5], *sw1 = br[b][6], *sb1 = br[b][7],
                *sw2 = br[b][8], *sb2 = br[b][9], *sw3 = br[b][10], *sb3 = br[b][11];
    k_prep_w1<<<D_IN, 256, 0, stream>>>(aw1, bg, bb, ab1, W1t, bias1);
    k_transpose_f16<<<D_IN, 256, 0, stream>>>(sw1, S1t);
    k_transpose_f16<<<D_IN, 256, 0, stream>>>(sw2, S2t);
    k_att<<<L_ROWS / MROWS, 256, 0, stream>>>(XN, W1t, bias1, aw2, ab2, attb);
    k_enc_partial<<<NCHUNK, 128, 0, stream>>>(attb, Q, csPart);
    k_enc_scan<<<1, 128, 0, stream>>>(csPart);
    k_enc_tin<<<NCHUNK, 128, 0, stream>>>(attb, Q, C, csPart, TIN);
    k_state<<<L_ROWS / MROWS, 256, 0, stream>>>(TIN, S1t, sb1, S2t, sb2, sw3, sb3, st[b]);
  }
  k_prob<<<L_ROWS / 256, 256, 0, stream>>>(st[0], st[1], user_init, Qinfo, cd_guess, out);
}
